// GCNModule_40776419508519
// MI455X (gfx1250) — matrix operand order repaired and checked by reading, not compiled, not run on hardware
//
#include <hip/hip_runtime.h>
#include <hip/hip_bf16.h>

typedef __attribute__((ext_vector_type(16))) __bf16 v16bf;
typedef __attribute__((ext_vector_type(8)))  float  v8f;
typedef __attribute__((ext_vector_type(4)))  float  v4f;

#define DFEAT 128
#define FRAG_ELEMS (8 * 4 * 32 * 16)   // 16384 bf16 per packed weight = 32 KiB

// ---------------------------------------------------------------------------
// Pack a 128x128 row-major f32 weight matrix into bf16 WMMA B-fragments.
// B-fragment layout used here (32x16, K x N; contiguous k, same order as A):
//   lane l: N = (l&15), half = l>>4, element e (0..15): K = kc*32 + 16*half + e
// Packed as P[((n*4 + kc)*32 + lane)*16 + e], n = N-tile (0..7), kc = K-chunk.
// ---------------------------------------------------------------------------
__global__ void __launch_bounds__(256)
pack_w_kernel(const float* __restrict__ W, __bf16* __restrict__ P)
{
    int t = blockIdx.x * 256 + threadIdx.x;      // 0 .. 1023
    if (t >= 8 * 4 * 32) return;
    int lane = t & 31;
    int kc   = (t >> 5) & 3;
    int n    = t >> 7;
    int half = lane >> 4;
    int N    = n * 16 + (lane & 15);
    __bf16* dst = P + (size_t)t * 16;
#pragma unroll
    for (int e = 0; e < 16; ++e) {
        int K = kc * 32 + 16 * half + e;
        dst[e] = (__bf16)W[K * DFEAT + N];
    }
}

// ---------------------------------------------------------------------------
// WMMA GEMM: C1 = A @ W1  (and, if DUAL, C2 = A @ W2 from the same A load).
// A: [M,128] f32 row-major (M multiple of 16). One wave -> 16x128 strip.
// Packed weights are staged in LDS once per block; inner loop feeds the
// matrix unit from ds_load_b128.
// ---------------------------------------------------------------------------
template <bool DUAL>
__global__ void __launch_bounds__(256)
wmma_gemm(const float*  __restrict__ A,
          const __bf16* __restrict__ Bp1,
          const __bf16* __restrict__ Bp2,
          float* __restrict__ C1,
          float* __restrict__ C2,
          int M)
{
    __shared__ __bf16 ldsW[(DUAL ? 2 : 1) * FRAG_ELEMS];

    // cooperative weight staging: 2048 (4096) uint4 across 256 threads
    {
        const uint4* s1 = (const uint4*)Bp1;
        uint4*       d1 = (uint4*)ldsW;
#pragma unroll
        for (int i = threadIdx.x; i < FRAG_ELEMS / 8; i += 256) d1[i] = s1[i];
        if (DUAL) {
            const uint4* s2 = (const uint4*)Bp2;
            uint4*       d2 = (uint4*)(ldsW + FRAG_ELEMS);
#pragma unroll
            for (int i = threadIdx.x; i < FRAG_ELEMS / 8; i += 256) d2[i] = s2[i];
        }
    }
    __syncthreads();

    const int lane   = threadIdx.x & 31;
    const int wave   = threadIdx.x >> 5;
    const int tile   = blockIdx.x * 8 + wave;    // 16-row tile index
    const int nTiles = M >> 4;
    if (tile >= nTiles) return;                  // wave-uniform exit (post-barrier)
    const int rowBase = tile << 4;
    const int half    = lane >> 4;
    const int mrow    = lane & 15;
    const float* arow = A + (size_t)(rowBase + mrow) * DFEAT;

    v8f acc1[8] = {};
    v8f acc2[8] = {};

#pragma unroll
    for (int kc = 0; kc < 4; ++kc) {
        // A fragment (16x32 bf16): element e -> K = kc*32 + 16*half + e
        const int k0 = kc * 32 + 16 * half;
        v4f a0 = *(const v4f*)(arow + k0);
        v4f a1 = *(const v4f*)(arow + k0 + 4);
        v4f a2 = *(const v4f*)(arow + k0 + 8);
        v4f a3 = *(const v4f*)(arow + k0 + 12);
        v16bf afrag;
#pragma unroll
        for (int e = 0; e < 4; ++e) {
            afrag[e]      = (__bf16)a0[e];
            afrag[4 + e]  = (__bf16)a1[e];
            afrag[8 + e]  = (__bf16)a2[e];
            afrag[12 + e] = (__bf16)a3[e];
        }
#pragma unroll
        for (int n = 0; n < 8; ++n) {
            const int fo = ((n * 4 + kc) * 32 + lane) * 16;
            v16bf b1 = *(const v16bf*)(ldsW + fo);
            acc1[n] = __builtin_amdgcn_wmma_f32_16x16x32_bf16(
                false, afrag, false, b1, (short)0, acc1[n], false, false);
            if (DUAL) {
                v16bf b2 = *(const v16bf*)(ldsW + FRAG_ELEMS + fo);
                acc2[n] = __builtin_amdgcn_wmma_f32_16x16x32_bf16(
                    false, afrag, false, b2, (short)0, acc2[n], false, false);
            }
        }
    }

    // C/D layout: VGPR r, lane l -> M = r + 8*(l>>4), N = (l&15)
#pragma unroll
    for (int n = 0; n < 8; ++n) {
        int col = n * 16 + mrow;
#pragma unroll
        for (int r = 0; r < 8; ++r) {
            size_t idx = (size_t)(rowBase + r + 8 * half) * DFEAT + col;
            C1[idx] = acc1[n][r];
            if (DUAL) C2[idx] = acc2[n][r];
        }
    }
}

// ---------------------------------------------------------------------------
// Graph helpers
// ---------------------------------------------------------------------------
__global__ void __launch_bounds__(256)
zero_kernel(float* __restrict__ p, int n)
{
    int i = blockIdx.x * 256 + threadIdx.x;
    if (i < n) p[i] = 0.0f;
}

__global__ void __launch_bounds__(256)
degree_kernel(const long long* __restrict__ dst, const float* __restrict__ w,
              float* __restrict__ deg, int E)
{
    int e = blockIdx.x * 256 + threadIdx.x;
    if (e < E) atomicAdd(&deg[dst[e]], w[e]);
}

__global__ void __launch_bounds__(256)
rsqrt_kernel(float* __restrict__ deg, int n)   // in place: dinv = rsqrt(deg+1)
{
    int i = blockIdx.x * 256 + threadIdx.x;
    if (i < n) deg[i] = rsqrtf(deg[i] + 1.0f);
}

// agg[i,:] = feat[i,:] * dinv[i]^2   (self-loop term, also initializes agg)
__global__ void __launch_bounds__(256)
selfloop_init_kernel(const float* __restrict__ feat, const float* __restrict__ dinv,
                     float* __restrict__ agg, int total)
{
    int i = blockIdx.x * 256 + threadIdx.x;
    if (i >= total) return;
    float dv = dinv[i >> 7];
    agg[i] = feat[i] * dv * dv;
}

// one wave per edge: agg[dst,:] += feat[src,:] * (dinv[src]*w*dinv[dst])
__global__ void __launch_bounds__(256)
edge_scatter_kernel(const float* __restrict__ feat,
                    const long long* __restrict__ src,
                    const long long* __restrict__ dst,
                    const float* __restrict__ w,
                    const float* __restrict__ dinv,
                    float* __restrict__ agg, int E)
{
    int e = blockIdx.x * 8 + (threadIdx.x >> 5);
    if (e >= E) return;
    int lane = threadIdx.x & 31;
    long long s = src[e];
    long long d = dst[e];
    float norm = dinv[s] * w[e] * dinv[d];
    v4f v = *(const v4f*)(feat + (size_t)s * DFEAT + lane * 4);
    float* ap = agg + (size_t)d * DFEAT + lane * 4;
    atomicAdd(ap + 0, v[0] * norm);
    atomicAdd(ap + 1, v[1] * norm);
    atomicAdd(ap + 2, v[2] * norm);
    atomicAdd(ap + 3, v[3] * norm);
}

__device__ __forceinline__ float gelu_exact(float x)
{
    return 0.5f * x * (1.0f + erff(x * 0.70710678118654752f));
}

// h = gelu(agg + b1) + 0.3*(xres + bres)
__global__ void __launch_bounds__(256)
epilogue1_kernel(const float* __restrict__ agg,  const float* __restrict__ b1,
                 const float* __restrict__ xres, const float* __restrict__ bres,
                 float* __restrict__ h, int total)
{
    int i = blockIdx.x * 256 + threadIdx.x;
    if (i >= total) return;
    int col = i & (DFEAT - 1);
    h[i] = gelu_exact(agg[i] + b1[col]) + 0.3f * (xres[i] + bres[col]);
}

// out = gelu(agg + b2)  (in place on d_out)
__global__ void __launch_bounds__(256)
epilogue2_kernel(float* __restrict__ agg, const float* __restrict__ b2, int total)
{
    int i = blockIdx.x * 256 + threadIdx.x;
    if (i >= total) return;
    int col = i & (DFEAT - 1);
    agg[i] = gelu_exact(agg[i] + b2[col]);
}

// ---------------------------------------------------------------------------
extern "C" void kernel_launch(void* const* d_in, const int* in_sizes, int n_in,
                              void* d_out, int out_size, void* d_ws, size_t ws_size,
                              hipStream_t stream)
{
    const float*     x    = (const float*)d_in[0];
    const long long* ei   = (const long long*)d_in[1];   // [2,E]
    const float*     w    = (const float*)d_in[5];
    const float*     W1   = (const float*)d_in[7];
    const float*     b1   = (const float*)d_in[8];
    const float*     W2   = (const float*)d_in[9];
    const float*     b2   = (const float*)d_in[10];
    const float*     Wres = (const float*)d_in[11];
    const float*     bres = (const float*)d_in[12];

    const int BN = in_sizes[0] / DFEAT;     // 100000
    const int E  = in_sizes[5];             // 1600000
    const long long* src = ei;
    const long long* dst = ei + E;

    // ---- workspace carve-up (all offsets 512B aligned) ----
    char* ws = (char*)d_ws;
    const size_t PACK_BYTES = (size_t)FRAG_ELEMS * sizeof(__bf16);   // 32 KiB
    __bf16* P1 = (__bf16*)(ws);
    __bf16* P2 = (__bf16*)(ws + PACK_BYTES);
    __bf16* Pr = (__bf16*)(ws + 2 * PACK_BYTES);
    size_t off = 3 * PACK_BYTES;
    off = (off + 511) & ~(size_t)511;
    float* dinv = (float*)(ws + off);
    off += ((size_t)BN * sizeof(float) + 511) & ~(size_t)511;
    float* bufA = (float*)(ws + off);                   // xw1, then h
    off += (size_t)BN * DFEAT * sizeof(float);
    float* bufB = (float*)(ws + off);                   // xres, then h@W2
    float* agg  = (float*)d_out;                        // aggregation buffer

    const int total   = BN * DFEAT;
    const int gElem   = (total + 255) / 256;
    const int gNode   = (BN + 255) / 256;
    const int gEdge1  = (E + 255) / 256;
    const int gEdge32 = (E + 7) / 8;
    const int gGemm   = ((BN >> 4) + 7) / 8;

    // 0) pack weights into WMMA-ready bf16 fragments
    pack_w_kernel<<<4, 256, 0, stream>>>(W1,   P1);
    pack_w_kernel<<<4, 256, 0, stream>>>(W2,   P2);
    pack_w_kernel<<<4, 256, 0, stream>>>(Wres, Pr);

    // 1) degrees -> dinv
    zero_kernel<<<gNode, 256, 0, stream>>>(dinv, BN);
    degree_kernel<<<gEdge1, 256, 0, stream>>>(dst, w, dinv, E);
    rsqrt_kernel<<<gNode, 256, 0, stream>>>(dinv, BN);

    // 2) layer 1 GEMMs (fused): xw1 = x@W1, xres = x@Wres
    wmma_gemm<true><<<gGemm, 256, 0, stream>>>(x, P1, Pr, bufA, bufB, BN);

    // 3) layer 1 aggregation
    selfloop_init_kernel<<<gElem, 256, 0, stream>>>(bufA, dinv, agg, total);
    edge_scatter_kernel<<<gEdge32, 256, 0, stream>>>(bufA, src, dst, w, dinv, agg, E);

    // 4) h = gelu(agg + b1) + 0.3*(xres + bres)
    epilogue1_kernel<<<gElem, 256, 0, stream>>>(agg, b1, bufB, bres, bufA, total);

    // 5) layer 2 GEMM: hw2 = h@W2
    wmma_gemm<false><<<gGemm, 256, 0, stream>>>(bufA, P2, nullptr, bufB, nullptr, BN);

    // 6) layer 2 aggregation
    selfloop_init_kernel<<<gElem, 256, 0, stream>>>(bufB, dinv, agg, total);
    edge_scatter_kernel<<<gEdge32, 256, 0, stream>>>(bufB, src, dst, w, dinv, agg, E);

    // 7) out = gelu(agg + b2)
    epilogue2_kernel<<<gElem, 256, 0, stream>>>(agg, b2, total);
}
